// RNN_438086664357
// MI455X (gfx1250) — matrix operand order repaired and checked by reading, not compiled, not run on hardware
//
#include <hip/hip_runtime.h>
#include <math.h>

// RNN scan: h_{t+1} = tanh(W [h_t; x_t] + b),  T=256, B=256, D_IN=D_H=1024.
// Persistent-kernel design: ONE launch runs all 256 timesteps; steps are
// separated by a device-scope atomic grid barrier (32 co-resident WGs).
// GEMM core: bf16 WMMA (v_wmma_f32_16x16x32_bf16), fp32 accumulate,
// double-buffered LDS staging, all-b128 LDS traffic.
// d_ws layout: [0, 4MB) = W in bf16;  at 4MB: u32 barrier counter.

typedef __attribute__((ext_vector_type(16))) __bf16 v16bf;
typedef __attribute__((ext_vector_type(8)))  float  v8f;

#define T_STEPS 256
#define BATCH   256
#define D_HH    1024
#define K_TOT   2048          // D_IN + D_H

#define TM  64                // batch rows per WG
#define TN  128               // d_h cols per WG
#define TK  32                // K chunk
#define LDA 40                // padded LDS stride (bf16), 80B: 16B-aligned + bank-skewed
#define LDB 40
#define NCHUNK (K_TOT / TK)   // 64
#define GRID_X (D_HH / TN)    // 8
#define GRID_Y (BATCH / TM)   // 4
#define NWG    (GRID_X * GRID_Y)  // 32 workgroups, all co-resident

union FragBF { v16bf v; uint4 q[2]; };
union Pack8  { __bf16 h[8]; uint4 q; };
union Pack4  { __bf16 h[4]; uint2 u; };

// ---- fp32 -> bf16 conversion of W (1024 x 2048), 4 elems/thread ----
__global__ __launch_bounds__(256)
void wcvt_kernel(const float* __restrict__ W, __bf16* __restrict__ Wb) {
    size_t i = ((size_t)blockIdx.x * 256 + threadIdx.x) * 4;
    float4 f = *(const float4*)(W + i);
    Pack4 p;
    p.h[0] = (__bf16)f.x; p.h[1] = (__bf16)f.y;
    p.h[2] = (__bf16)f.z; p.h[3] = (__bf16)f.w;
    *(uint2*)(Wb + i) = p.u;
}

// ---- zero h0 slab of the output ----
__global__ __launch_bounds__(256)
void zero_kernel(float* __restrict__ p) {
    size_t i = ((size_t)blockIdx.x * 256 + threadIdx.x) * 4;
    float4 z = {0.f, 0.f, 0.f, 0.f};
    *(float4*)(p + i) = z;
}

// ---- zero the grid-barrier counter (must happen every call / replay) ----
__global__ void zcnt_kernel(unsigned* __restrict__ cnt) { *cnt = 0u; }

// stage one K-chunk: A tile (fp32->bf16) and B^T tile (bf16), all b128
__device__ __forceinline__
void stage_chunk(__bf16* __restrict__ Ab, __bf16* __restrict__ Bb,
                 const float* __restrict__ ap, const __bf16* __restrict__ bp,
                 int ar, int aq, int br, int bq) {
    float4 f0 = *(const float4*)ap;
    float4 f1 = *(const float4*)(ap + 4);
    Pack8 pk;
    pk.h[0] = (__bf16)f0.x; pk.h[1] = (__bf16)f0.y;
    pk.h[2] = (__bf16)f0.z; pk.h[3] = (__bf16)f0.w;
    pk.h[4] = (__bf16)f1.x; pk.h[5] = (__bf16)f1.y;
    pk.h[6] = (__bf16)f1.z; pk.h[7] = (__bf16)f1.w;
    *(uint4*)&Ab[ar * LDA + aq]     = pk.q;
    *(uint4*)&Bb[br * LDB + bq]     = *(const uint4*)bp;
    *(uint4*)&Bb[br * LDB + bq + 8] = *(const uint4*)(bp + 8);
}

__global__ __launch_bounds__(256)
void rnn_persistent(const float*  __restrict__ seq,   // [T][B][D_H]
                    const __bf16* __restrict__ Wb,    // [D_H][K_TOT]
                    const float*  __restrict__ bias,  // [D_H]
                    float*        __restrict__ out,   // [T+1][B][D_H]
                    unsigned*     __restrict__ cnt)   // grid barrier counter
{
    __shared__ __bf16 As[2][TM * LDA];   // 2 x 5120 B
    __shared__ __bf16 Bs[2][TN * LDB];   // 2 x 10240 B

    const int tid  = threadIdx.x;
    const int lane = tid & 31;
    const int half = lane >> 4;
    const int l16  = lane & 15;
    const int wid  = tid >> 5;

    const int mBase = blockIdx.y * TM;
    const int nBase = blockIdx.x * TN;
    const int mWave = (wid >> 2) * 32;
    const int nWave = (wid & 3)  * 32;

    // LDS fill assignments
    const int ar = tid >> 2;          // A row 0..63
    const int aq = (tid & 3) * 8;     // A col {0,8,16,24}
    const int br = tid >> 1;          // B row (n) 0..127
    const int bq = (tid & 1) * 16;    // B k {0,16}

    // per-lane fragment offsets (CDNA5 16-bit WMMA swizzle)
    const int aoff0 = (mWave + l16) * LDA + half * 8;
    const int aoff1 = aoff0 + 16 * LDA;
    const int boff0 = (nWave + l16) * LDB + half * 8;
    const int boff1 = boff0 + 16 * LDB;

    // time-invariant pieces
    const __bf16* wbase = Wb + (size_t)(nBase + br) * K_TOT + bq;
    const int n0 = nBase + nWave + l16;
    const int n1 = n0 + 16;
    const float bias0 = bias[n0];
    const float bias1 = bias[n1];
    const int mRow = mBase + mWave + half * 8;

#pragma clang loop unroll(disable)
    for (int t = 0; t < T_STEPS; ++t) {
        const float* hprev = out + (size_t)t * BATCH * D_HH;
        const float* xt    = seq + (size_t)t * BATCH * D_HH;
        float*       hnext = out + (size_t)(t + 1) * BATCH * D_HH;
        const float* aH = hprev + (size_t)(mBase + ar) * D_HH + aq;
        const float* aX = xt    + (size_t)(mBase + ar) * D_HH + aq;

        v8f acc00 = {}, acc01 = {}, acc10 = {}, acc11 = {};

        // prologue: chunk 0 into buffer 0 (K 0..31 multiplies h_t)
        stage_chunk(&As[0][0], &Bs[0][0], aH, wbase, ar, aq, br, bq);

        for (int kc = 0; kc < NCHUNK; ++kc) {
            const int cur = kc & 1;
            __syncthreads();   // fills of buf[cur] visible; reads of buf[!cur] done
            if (kc + 1 < NCHUNK) {
                const int kn = kc + 1;
                const float* ap = (kn < 32) ? (aH + kn * TK) : (aX + (kn - 32) * TK);
                stage_chunk(&As[cur ^ 1][0], &Bs[cur ^ 1][0],
                            ap, wbase + kn * TK, ar, aq, br, bq);
            }
            const __bf16* Ab = &As[cur][0];
            const __bf16* Bb = &Bs[cur][0];
            FragBF a0, a1, b0, b1;
            a0.q[0] = *(const uint4*)&Ab[aoff0];
            a0.q[1] = *(const uint4*)&Ab[aoff0 + 16];
            a1.q[0] = *(const uint4*)&Ab[aoff1];
            a1.q[1] = *(const uint4*)&Ab[aoff1 + 16];
            b0.q[0] = *(const uint4*)&Bb[boff0];
            b0.q[1] = *(const uint4*)&Bb[boff0 + 16];
            b1.q[0] = *(const uint4*)&Bb[boff1];
            b1.q[1] = *(const uint4*)&Bb[boff1 + 16];
            acc00 = __builtin_amdgcn_wmma_f32_16x16x32_bf16(false, a0.v, false, b0.v, (short)0, acc00, false, false);
            acc01 = __builtin_amdgcn_wmma_f32_16x16x32_bf16(false, a0.v, false, b1.v, (short)0, acc01, false, false);
            acc10 = __builtin_amdgcn_wmma_f32_16x16x32_bf16(false, a1.v, false, b0.v, (short)0, acc10, false, false);
            acc11 = __builtin_amdgcn_wmma_f32_16x16x32_bf16(false, a1.v, false, b1.v, (short)0, acc11, false, false);
        }

        // epilogue: bias + tanh, fp32 stores of h_{t+1}
#pragma unroll
        for (int e = 0; e < 8; ++e) {
            const int m0 = mRow + e;
            const int m1 = m0 + 16;
            hnext[(size_t)m0 * D_HH + n0] = tanhf(acc00[e] + bias0);
            hnext[(size_t)m0 * D_HH + n1] = tanhf(acc01[e] + bias1);
            hnext[(size_t)m1 * D_HH + n0] = tanhf(acc10[e] + bias0);
            hnext[(size_t)m1 * D_HH + n1] = tanhf(acc11[e] + bias1);
        }

        // ---- device-wide barrier: h_{t+1} must be visible to all WGs ----
        __syncthreads();
        if (tid == 0) {
            __threadfence();  // release h_{t+1} stores (agent scope)
            __hip_atomic_fetch_add(cnt, 1u, __ATOMIC_RELEASE, __HIP_MEMORY_SCOPE_AGENT);
            const unsigned target = (unsigned)(t + 1) * NWG;
            while (__hip_atomic_load(cnt, __ATOMIC_ACQUIRE, __HIP_MEMORY_SCOPE_AGENT) < target)
                __builtin_amdgcn_s_sleep(2);
        }
        __syncthreads();
        __threadfence();      // acquire: invalidate stale lines before reading h_{t+1}
    }
}

extern "C" void kernel_launch(void* const* d_in, const int* in_sizes, int n_in,
                              void* d_out, int out_size, void* d_ws, size_t ws_size,
                              hipStream_t stream) {
    (void)in_sizes; (void)n_in; (void)out_size; (void)ws_size;
    const float* seq = (const float*)d_in[0];
    const float* W   = (const float*)d_in[1];
    const float* b   = (const float*)d_in[2];
    float* out = (float*)d_out;
    __bf16*   Wb  = (__bf16*)d_ws;                              // 4 MB
    unsigned* cnt = (unsigned*)((char*)d_ws + (4u << 20));      // +4 bytes

    // 1) W fp32 -> bf16
    wcvt_kernel<<<(D_HH * K_TOT) / (256 * 4), 256, 0, stream>>>(W, Wb);
    // 2) zero h0 slab
    zero_kernel<<<(BATCH * D_HH) / (256 * 4), 256, 0, stream>>>(out);
    // 3) reset grid-barrier counter (every call -> deterministic replays)
    zcnt_kernel<<<1, 1, 0, stream>>>(cnt);
    // 4) persistent scan kernel: one launch for all 256 steps
    dim3 grid(GRID_X, GRID_Y);   // 32 WGs, all co-resident
    rnn_persistent<<<grid, 256, 0, stream>>>(seq, Wb, b, out, cnt);
}
